// ISA_79843442033234
// MI455X (gfx1250) — matrix operand order repaired and checked by reading, not compiled, not run on hardware
//
#include <hip/hip_runtime.h>

// ---------------------------------------------------------------------------
// Transposed (channel) attention for b=2, c=192, n=256*256, fused for MI455X:
//   qkvT = Wqkv @ x  (bf16, WMMA)  -> q,k,v in [c, n] layout
//   norms of q,k rows (folded into score scaling)
//   S = q @ k^T      (bf16 WMMA, K=65536; double-buffered async global->LDS
//                     staging when the gfx1250 async builtins are available)
//   softmax(S * temp / (|q||k|))
//   M = proj_w @ S   (tiny fp32)
//   out = M @ v + proj_b  (bf16 WMMA, fp32 out)
// ---------------------------------------------------------------------------

#define B_    2
#define C_    192
#define C3_   576
#define N_    65536
#define LDSP  200   // padded LDS row stride (bf16 elems) for 64-token tiles

typedef __attribute__((ext_vector_type(16))) __bf16       v16bf;
typedef __attribute__((ext_vector_type(8)))  float        v8f;
typedef __attribute__((ext_vector_type(4)))  unsigned int u32x4;
typedef __attribute__((ext_vector_type(4)))  int          i32x4;
typedef __attribute__((ext_vector_type(4)))  float        f32x4;

union Frag  { v16bf v; u32x4 q[2]; unsigned short u[16]; };
union Pack8 { u32x4 q; unsigned short s[8]; };

__device__ __forceinline__ unsigned short f2bf(float f) {
    unsigned int u = __float_as_uint(f);
    u += 0x7fffu + ((u >> 16) & 1u);           // round-to-nearest-even
    return (unsigned short)(u >> 16);
}
__device__ __forceinline__ float bf2f(unsigned short h) {
    return __uint_as_float(((unsigned int)h) << 16);
}
__device__ __forceinline__ v8f vzero8() {
    v8f z;
    for (int i = 0; i < 8; ++i) z[i] = 0.0f;
    return z;
}

#if __has_builtin(__builtin_amdgcn_global_load_async_to_lds_b128) && \
    __has_builtin(__builtin_amdgcn_s_wait_asynccnt)
#define HAVE_GFX1250_ASYNC 1
#else
#define HAVE_GFX1250_ASYNC 0
#endif

#if HAVE_GFX1250_ASYNC
// One 16B-per-lane async block copy, global -> LDS (tracked by ASYNCcnt).
// Probe-derived signature: (AS(1) int4*, AS(3) int4*, imm offset, imm cpol).
typedef __attribute__((address_space(1))) i32x4* gv4i_p;
typedef __attribute__((address_space(3))) i32x4* lv4i_p;
__device__ __forceinline__ void async_cp16(const unsigned short* g,
                                           unsigned short* l) {
    __builtin_amdgcn_global_load_async_to_lds_b128(
        (gv4i_p)(unsigned short*)g, (lv4i_p)l, 0, 0);
}
#endif

// ---------------------------------------------------------------------------
// Kernel 1: fp32 -> bf16 weight conversion (qkv_w, 576x192)
// ---------------------------------------------------------------------------
__global__ void k_cvt(const float* __restrict__ src,
                      unsigned short* __restrict__ dst, int n) {
    int i = blockIdx.x * 256 + threadIdx.x;
    if (i < n) dst[i] = f2bf(src[i]);
}

// ---------------------------------------------------------------------------
// Kernel 2: qkvT[b] = Wqkv[576,192] @ x[b][192,N] + bias, bf16 output.
// Block = 128 thr (4 waves). Tile: M=64 (16 per wave), N=64, full K=192.
// x tile transposed through LDS with fp32->bf16 conversion.
// ---------------------------------------------------------------------------
__global__ __launch_bounds__(128) void k_gemm_qkv(
    const float* __restrict__ x, const unsigned short* __restrict__ wq,
    const float* __restrict__ qkv_b, unsigned short* __restrict__ qkvT)
{
    __shared__ unsigned short sB[64 * LDSP];
    const int t = threadIdx.x, wv = t >> 5, lane = t & 31;
    const int lm = lane & 15, lh = lane >> 4;
    const int n0 = blockIdx.x * 64, m0 = blockIdx.y * 64, b = blockIdx.z;
    const float* xb = x + (size_t)b * C_ * N_;

    // Stage 64 tokens x 192 channels, transposed to [token][channel] bf16.
    for (int r = 0; r < 24; ++r) {
        int lin = r * 128 + t;
        int ch = lin >> 4, t4 = (lin & 15) * 4;
        const f32x4 f = *(const f32x4*)(xb + (size_t)ch * N_ + n0 + t4);
        sB[(t4 + 0) * LDSP + ch] = f2bf(f.x);
        sB[(t4 + 1) * LDSP + ch] = f2bf(f.y);
        sB[(t4 + 2) * LDSP + ch] = f2bf(f.z);
        sB[(t4 + 3) * LDSP + ch] = f2bf(f.w);
    }
    __syncthreads();

    v8f acc[4];
    for (int s = 0; s < 4; ++s) acc[s] = vzero8();

    const unsigned short* Ap = wq + (m0 + wv * 16 + lm) * C_;  // K contiguous
    for (int kk = 0; kk < C_; kk += 32) {
        Frag a;
        a.q[0] = *(const u32x4*)(Ap + kk + lh * 8);        // K = kk+lh*8 .. +7
        a.q[1] = *(const u32x4*)(Ap + kk + lh * 8 + 16);   // K = +16
        for (int s = 0; s < 4; ++s) {
            Frag bb;
            const unsigned short* Bp = &sB[(s * 16 + lm) * LDSP + kk + lh * 8];
            bb.q[0] = *(const u32x4*)(Bp);
            bb.q[1] = *(const u32x4*)(Bp + 16);
            acc[s] = __builtin_amdgcn_wmma_f32_16x16x32_bf16(
                false, a.v, false, bb.v, (short)0, acc[s], false, false);
        }
    }

    for (int s = 0; s < 4; ++s)
        for (int r = 0; r < 8; ++r) {
            int row = m0 + wv * 16 + lh * 8 + r;
            int col = n0 + s * 16 + lm;
            qkvT[((size_t)b * C3_ + row) * N_ + col] = f2bf(acc[s][r] + qkv_b[row]);
        }
}

// ---------------------------------------------------------------------------
// Kernel 3: L2 norms of the first 2c rows (q then k) of qkvT, per batch.
// ---------------------------------------------------------------------------
__global__ __launch_bounds__(256) void k_rownorm(
    const unsigned short* __restrict__ qkvT, float* __restrict__ norms)
{
    __shared__ float red[256];
    const int row = blockIdx.x;                 // 0 .. B*2C-1
    const int b = row / (2 * C_), ch = row % (2 * C_);
    const unsigned short* p = qkvT + ((size_t)b * C3_ + ch) * N_;
    float s = 0.f;
    for (int i = threadIdx.x; i < N_ / 8; i += 256) {
        Pack8 u; u.q = *(const u32x4*)(p + i * 8);
        for (int j = 0; j < 8; ++j) { float v = bf2f(u.s[j]); s += v * v; }
    }
    red[threadIdx.x] = s;
    __syncthreads();
    for (int st = 128; st > 0; st >>= 1) {
        if (threadIdx.x < st) red[threadIdx.x] += red[threadIdx.x + st];
        __syncthreads();
    }
    if (threadIdx.x == 0) norms[row] = sqrtf(red[0]);
}

// ---------------------------------------------------------------------------
// Kernel 4: S[b][c,d] = sum_n q[c,n] k[d,n].  K=65536 split over 4 waves.
// Async path: per-wave double-buffered global->LDS block copies (ASYNCcnt),
// WMMA fragments then come from LDS.  Fallback: direct-from-global fragments.
// ---------------------------------------------------------------------------
__global__ __launch_bounds__(128) void k_scores(
    const unsigned short* __restrict__ qkvT, float* __restrict__ S)
{
    __shared__ float red[4 * 256];
    const int t = threadIdx.x, wv = t >> 5, lane = t & 31;
    const int lm = lane & 15, lh = lane >> 4;
    const int m0 = blockIdx.x * 16, n0 = blockIdx.y * 16, b = blockIdx.z;

    v8f acc = vzero8();

#if HAVE_GFX1250_ASYNC
    // [wave][buffer][q/k][16 rows x 64 K-elems] bf16 = 32 KB total.
    __shared__ unsigned short stg[4][2][2][16 * 64];
    const unsigned short* qbase = qkvT + (size_t)b * C3_ * N_;
    const int crow = lane >> 3;            // 4 rows per copy instruction
    const int ccol = (lane & 7) * 8;       // 8 x 16B chunks per row
    const int k0 = wv * (N_ / 4);
    const int NCH = (N_ / 4) / 64;         // 256 chunks of K=64 per wave

    // Prime buffer 0 (8 async copies: 4 for q rows, 4 for k rows).
    for (int j = 0; j < 4; ++j) {
        int row = j * 4 + crow;
        async_cp16(qbase + (size_t)(m0 + row) * N_ + k0 + ccol,
                   &stg[wv][0][0][row * 64 + ccol]);
        async_cp16(qbase + (size_t)(C_ + n0 + row) * N_ + k0 + ccol,
                   &stg[wv][0][1][row * 64 + ccol]);
    }
    for (int c = 0; c < NCH; ++c) {
        const int buf = c & 1, kc = k0 + c * 64;
        if (c + 1 < NCH) {
            const int nb = buf ^ 1, kn = kc + 64;
            for (int j = 0; j < 4; ++j) {
                int row = j * 4 + crow;
                async_cp16(qbase + (size_t)(m0 + row) * N_ + kn + ccol,
                           &stg[wv][nb][0][row * 64 + ccol]);
                async_cp16(qbase + (size_t)(C_ + n0 + row) * N_ + kn + ccol,
                           &stg[wv][nb][1][row * 64 + ccol]);
            }
            // 8 newer copies may remain in flight; the 8 for `buf` are done.
            __builtin_amdgcn_s_wait_asynccnt(8);
        } else {
            __builtin_amdgcn_s_wait_asynccnt(0);
        }
        for (int ks = 0; ks < 2; ++ks) {
            Frag a, bb;
            const unsigned short* Ap = &stg[wv][buf][0][lm * 64 + ks * 32];
            const unsigned short* Bp = &stg[wv][buf][1][lm * 64 + ks * 32];
            a.q[0]  = *(const u32x4*)(Ap + lh * 8);
            a.q[1]  = *(const u32x4*)(Ap + lh * 8 + 16);
            bb.q[0] = *(const u32x4*)(Bp + lh * 8);
            bb.q[1] = *(const u32x4*)(Bp + lh * 8 + 16);
            acc = __builtin_amdgcn_wmma_f32_16x16x32_bf16(
                false, a.v, false, bb.v, (short)0, acc, false, false);
        }
    }
#else
    const unsigned short* qp = qkvT + ((size_t)b * C3_ + m0 + lm) * N_;
    const unsigned short* kp = qkvT + ((size_t)b * C3_ + C_ + n0 + lm) * N_;
    const int k0 = wv * (N_ / 4), k1 = k0 + N_ / 4;
    for (int k = k0; k < k1; k += 32) {
        __builtin_prefetch(qp + k + 1024, 0, 1);   // global_prefetch_b8
        __builtin_prefetch(kp + k + 1024, 0, 1);
        Frag a, bb;
        a.q[0]  = *(const u32x4*)(qp + k + lh * 8);
        a.q[1]  = *(const u32x4*)(qp + k + lh * 8 + 16);
        bb.q[0] = *(const u32x4*)(kp + k + lh * 8);
        bb.q[1] = *(const u32x4*)(kp + k + lh * 8 + 16);
        acc = __builtin_amdgcn_wmma_f32_16x16x32_bf16(
            false, a.v, false, bb.v, (short)0, acc, false, false);
    }
#endif

    for (int r = 0; r < 8; ++r) red[wv * 256 + lane * 8 + r] = acc[r];
    __syncthreads();
    if (wv == 0)
        for (int r = 0; r < 8; ++r) {
            float v = red[lane * 8 + r] + red[256 + lane * 8 + r] +
                      red[512 + lane * 8 + r] + red[768 + lane * 8 + r];
            int row = m0 + lh * 8 + r, col = n0 + lm;
            S[((size_t)b * C_ + row) * C_ + col] = v;
        }
}

// ---------------------------------------------------------------------------
// Kernel 5: scale by temp/(max(|q|,eps)*max(|k|,eps)) + row softmax (fp32).
// ---------------------------------------------------------------------------
__global__ __launch_bounds__(256) void k_softmax(
    float* __restrict__ S, const float* __restrict__ norms,
    const float* __restrict__ temperature)
{
    __shared__ float red[256];
    const int b = blockIdx.x / C_, c = blockIdx.x % C_;
    const int d = threadIdx.x;
    float* row = S + ((size_t)b * C_ + c) * C_;
    const float* nb = norms + b * (2 * C_);
    const float temp = temperature[0];
    const float qn = fmaxf(nb[c], 1e-12f);

    float val = 0.f, m0 = -INFINITY;
    if (d < C_) {
        float kn = fmaxf(nb[C_ + d], 1e-12f);
        val = row[d] * temp / (qn * kn);
        m0 = val;
    }
    red[d] = m0; __syncthreads();
    for (int st = 128; st > 0; st >>= 1) {
        if (d < st) red[d] = fmaxf(red[d], red[d + st]);
        __syncthreads();
    }
    const float mx = red[0]; __syncthreads();
    float e = (d < C_) ? __expf(val - mx) : 0.f;
    red[d] = e; __syncthreads();
    for (int st = 128; st > 0; st >>= 1) {
        if (d < st) red[d] += red[d + st];
        __syncthreads();
    }
    if (d < C_) row[d] = e / red[0];
}

// ---------------------------------------------------------------------------
// Kernel 6: W[b] = proj_w @ attn[b]  (192x192x192, tiny -> fp32 VALU), bf16 out
// ---------------------------------------------------------------------------
__global__ __launch_bounds__(256) void k_projattn(
    const float* __restrict__ proj_w, const float* __restrict__ S,
    unsigned short* __restrict__ W)
{
    int i = blockIdx.x * 256 + threadIdx.x;
    if (i >= B_ * C_ * C_) return;
    int b = i / (C_ * C_), rem = i % (C_ * C_);
    int dout = rem / C_, d = rem % C_;
    const float* pw = proj_w + dout * C_;
    const float* a  = S + (size_t)b * C_ * C_;
    float s = 0.f;
    for (int c = 0; c < C_; ++c) s += pw[c] * a[c * C_ + d];
    W[i] = f2bf(s);
}

// ---------------------------------------------------------------------------
// Kernel 7: out[b] = W[b] @ v[b] + proj_b, fp32 output in [b,c,h,w] layout.
// ---------------------------------------------------------------------------
__global__ __launch_bounds__(128) void k_gemm_out(
    const unsigned short* __restrict__ qkvT, const unsigned short* __restrict__ W,
    const float* __restrict__ proj_b, float* __restrict__ out)
{
    __shared__ unsigned short sB[64 * LDSP];
    const int t = threadIdx.x, wv = t >> 5, lane = t & 31;
    const int lm = lane & 15, lh = lane >> 4;
    const int n0 = blockIdx.x * 64, m0 = blockIdx.y * 64, b = blockIdx.z;
    const unsigned short* vb = qkvT + ((size_t)b * C3_ + 2 * C_) * N_;

    for (int r = 0; r < 12; ++r) {
        int lin = r * 128 + t;
        int ch = lin >> 3, t8 = (lin & 7) * 8;
        Pack8 u; u.q = *(const u32x4*)(vb + (size_t)ch * N_ + n0 + t8);
        for (int j = 0; j < 8; ++j) sB[(t8 + j) * LDSP + ch] = u.s[j];
    }
    __syncthreads();

    v8f acc[4];
    for (int s = 0; s < 4; ++s) acc[s] = vzero8();

    const unsigned short* Ap = W + (size_t)b * C_ * C_ + (m0 + wv * 16 + lm) * C_;
    for (int kk = 0; kk < C_; kk += 32) {
        Frag a;
        a.q[0] = *(const u32x4*)(Ap + kk + lh * 8);
        a.q[1] = *(const u32x4*)(Ap + kk + lh * 8 + 16);
        for (int s = 0; s < 4; ++s) {
            Frag bb;
            const unsigned short* Bp = &sB[(s * 16 + lm) * LDSP + kk + lh * 8];
            bb.q[0] = *(const u32x4*)(Bp);
            bb.q[1] = *(const u32x4*)(Bp + 16);
            acc[s] = __builtin_amdgcn_wmma_f32_16x16x32_bf16(
                false, a.v, false, bb.v, (short)0, acc[s], false, false);
        }
    }

    for (int s = 0; s < 4; ++s)
        for (int r = 0; r < 8; ++r) {
            int row = m0 + wv * 16 + lh * 8 + r;
            int col = n0 + s * 16 + lm;
            out[((size_t)b * C_ + row) * N_ + col] = acc[s][r] + proj_b[row];
        }
}

// ---------------------------------------------------------------------------
extern "C" void kernel_launch(void* const* d_in, const int* in_sizes, int n_in,
                              void* d_out, int out_size, void* d_ws, size_t ws_size,
                              hipStream_t stream)
{
    const float* x      = (const float*)d_in[0];
    const float* temp   = (const float*)d_in[1];
    const float* qkv_w  = (const float*)d_in[2];
    const float* qkv_b  = (const float*)d_in[3];
    const float* proj_w = (const float*)d_in[4];
    const float* proj_b = (const float*)d_in[5];
    float* out = (float*)d_out;

    char* ws = (char*)d_ws;
    size_t off = 0;
    unsigned short* qkvT = (unsigned short*)(ws + off); off += (size_t)B_ * C3_ * N_ * 2; // 151 MB
    unsigned short* wq   = (unsigned short*)(ws + off); off += (size_t)C3_ * C_ * 2;
    float*          nrm  = (float*)(ws + off);          off += (size_t)B_ * 2 * C_ * 4;
    float*          S    = (float*)(ws + off);          off += (size_t)B_ * C_ * C_ * 4;
    unsigned short* W    = (unsigned short*)(ws + off);

    k_cvt<<<(C3_ * C_ + 255) / 256, 256, 0, stream>>>(qkv_w, wq, C3_ * C_);
    k_gemm_qkv<<<dim3(N_ / 64, C3_ / 64, B_), 128, 0, stream>>>(x, wq, qkv_b, qkvT);
    k_rownorm<<<B_ * 2 * C_, 256, 0, stream>>>(qkvT, nrm);
    k_scores<<<dim3(C_ / 16, C_ / 16, B_), 128, 0, stream>>>(qkvT, S);
    k_softmax<<<B_ * C_, 256, 0, stream>>>(S, nrm, temp);
    k_projattn<<<(B_ * C_ * C_ + 255) / 256, 256, 0, stream>>>(proj_w, S, W);
    k_gemm_out<<<dim3(N_ / 64, C_ / 64, B_), 128, 0, stream>>>(qkvT, W, proj_b, out);
}
